// EnhancedMambaMixer_66065186947609
// MI455X (gfx1250) — matrix operand order repaired and checked by reading, not compiled, not run on hardware
//
#include <hip/hip_runtime.h>
#include <hip/hip_bf16.h>

// ---------------------------------------------------------------------------
// Mamba mixer forward for MI455X (gfx1250), wave32 + WMMA bf16 path.
// GEMMs:  v_wmma_f32_16x16x32_bf16, fragments per CDNA5 ISA 7.12.2 layouts.
// Scan:   fused exp/state/reduce/gate kernel, double-buffered async-to-LDS
//         staging (GLOBAL_LOAD_ASYNC_TO_LDS_B128 + s_wait_asynccnt) so the
//         sequential 1024-step chain runs entirely out of LDS.
// ---------------------------------------------------------------------------

typedef __attribute__((ext_vector_type(16))) __bf16         v16bf;
typedef __attribute__((ext_vector_type(8)))  float          v8f;
typedef __attribute__((ext_vector_type(4)))  float          v4f;
typedef __attribute__((ext_vector_type(4)))  int            v4i;
typedef __attribute__((ext_vector_type(16))) unsigned short v16u;
typedef __attribute__((ext_vector_type(8)))  unsigned short v8u;

constexpr int BB = 2, LL = 1024, HH = 1024, DD = 2048, NN = 16, KK = 4, RR = 64;
constexpr int MROWS = BB * LL;      // 2048 token rows
constexpr int SSMW  = RR + 2 * NN;  // 96
constexpr int CHUNK = 64;           // scan steps staged per LDS chunk
constexpr int NCHUNK = LL / CHUNK;  // 16

#if defined(__gfx1250__) && __has_builtin(__builtin_amdgcn_global_load_async_to_lds_b128) && __has_builtin(__builtin_amdgcn_s_wait_asynccnt)
#define USE_ASYNC_LDS 1
#else
#define USE_ASYNC_LDS 0
#endif

static __device__ __forceinline__ unsigned short f2bf(float f) {
  unsigned int u = __builtin_bit_cast(unsigned int, f);
  u += 0x7FFFu + ((u >> 16) & 1u);  // round-to-nearest-even
  return (unsigned short)(u >> 16);
}

// 16-byte global -> LDS copy: async DMA when available, VGPR bounce otherwise.
static __device__ __forceinline__ void cp16_g2l(const float* g, float* l) {
#if USE_ASYNC_LDS
  typedef __attribute__((address_space(1))) v4i* gp_t;  // global int4*
  typedef __attribute__((address_space(3))) v4i* lp_t;  // LDS int4* (32-bit)
  __builtin_amdgcn_global_load_async_to_lds_b128((gp_t)g, (lp_t)l, 0, 0);
#else
  *(v4f*)l = *(const v4f*)g;
#endif
}

static __device__ __forceinline__ void wait_chunk_ready() {
#if USE_ASYNC_LDS
  __builtin_amdgcn_s_wait_asynccnt(5);  // 5 issues/chunk; in-order completion
#endif
  __syncthreads();
}

// ---------------------------------------------------------------------------
// fp32 -> bf16 conversion
// ---------------------------------------------------------------------------
__global__ __launch_bounds__(256) void cvt_bf16_kernel(const float* __restrict__ in,
                                                       unsigned short* __restrict__ out,
                                                       int n) {
  int i = blockIdx.x * 256 + threadIdx.x;
  if (i < n) out[i] = f2bf(in[i]);
}

// ---------------------------------------------------------------------------
// C(MxN,f32) = A(MxK,bf16 row-major) * B(NxK,bf16 row-major)^T
// One wave -> 16 x (NT*16) strip, NT accumulators share one A fragment.
// The 4 waves of a block cover 4 consecutive M-tiles of the SAME N-group,
// so B fragments are block-shared and served out of L0/WGP$.
// Requires: M % 64 == 0, N % (NT*16) == 0, K % 32 == 0 (true for all uses).
// ---------------------------------------------------------------------------
template <int NT>
__global__ __launch_bounds__(128) void wmma_gemm_bf16_kernel(
    const unsigned short* __restrict__ A, const unsigned short* __restrict__ B,
    float* __restrict__ C, int M, int N, int Kd) {
  const int lane = threadIdx.x & 31;
  const int wave = threadIdx.x >> 5;
  const int c16  = lane & 15;   // A: row M | B: col N | C: col N
  const int half = lane >> 4;   // K half-select / C row half-select
  const int ngroups = N / (NT * 16);
  const int bt = blockIdx.x;
  const int nt = bt % ngroups;            // N-group, shared by all 4 waves
  const int mt = (bt / ngroups) * 4 + wave;
  const int m0 = mt * 16;
  const int n0 = nt * (NT * 16);
  (void)M;

  v8f acc[NT] = {};

  const unsigned short* arow  = A + (size_t)(m0 + c16) * Kd + 8 * half;
  const unsigned short* brow0 = B + (size_t)(n0 + c16) * Kd + 8 * half;

  for (int k0 = 0; k0 < Kd; k0 += 32) {
    // A 16x32 fragment: lane holds K in [8*half, +8) and [16+8*half, +8)
    v8u alo = *(const v8u*)(arow + k0);
    v8u ahi = *(const v8u*)(arow + k0 + 16);
    // prefetch the A stream two K-blocks ahead (global_prefetch_b8)
    __builtin_prefetch(arow + k0 + 64, 0, 1);
    v16u au;
#pragma unroll
    for (int i = 0; i < 8; ++i) { au[i] = alo[i]; au[i + 8] = ahi[i]; }
    v16bf af = __builtin_bit_cast(v16bf, au);

#pragma unroll
    for (int j = 0; j < NT; ++j) {
      // B 32x16 fragment: lane = column n, K in [8*half, +8) and [16+8*half, +8)
      const unsigned short* brow = brow0 + (size_t)(j * 16) * Kd + k0;
      v8u blo = *(const v8u*)brow;
      v8u bhi = *(const v8u*)(brow + 16);
      v16u bu;
#pragma unroll
      for (int i = 0; i < 8; ++i) { bu[i] = blo[i]; bu[i + 8] = bhi[i]; }
      v16bf bfr = __builtin_bit_cast(v16bf, bu);
      acc[j] = __builtin_amdgcn_wmma_f32_16x16x32_bf16(
          false, af, false, bfr, (short)0, acc[j], false, false);
    }
  }

  // C/D 16x16 f32: VGPR r -> row (8*half + r), lane&15 -> column
#pragma unroll
  for (int j = 0; j < NT; ++j) {
    float* cp = C + (size_t)(m0 + 8 * half) * N + n0 + j * 16 + c16;
#pragma unroll
    for (int r = 0; r < 8; ++r) cp[(size_t)r * N] = acc[j][r];
  }
}

// ---------------------------------------------------------------------------
// Depthwise causal conv (K=4) + bias + SiLU on the x half of proj.
// Writes f32 (for scan) and bf16 (for the ssm-proj WMMA GEMM).
// ---------------------------------------------------------------------------
__global__ __launch_bounds__(256) void conv_silu_kernel(
    const float* __restrict__ proj, const float* __restrict__ cw,
    const float* __restrict__ cb, float* __restrict__ xs,
    unsigned short* __restrict__ xbf) {
  int idx = blockIdx.x * 256 + threadIdx.x;
  if (idx >= MROWS * DD) return;
  int d = idx & (DD - 1);
  int m = idx >> 11;        // / DD
  int l = m & (LL - 1);
  float acc = cb[d];
#pragma unroll
  for (int k = 0; k < KK; ++k) {
    int ls = l - (KK - 1) + k;
    if (ls >= 0)
      acc += proj[(size_t)(m - (KK - 1) + k) * (2 * DD) + d] * cw[d * KK + k];
  }
  float s = acc / (1.f + __expf(-acc));  // silu
  xs[idx]  = s;
  xbf[idx] = f2bf(s);
}

// ---------------------------------------------------------------------------
// Pack dt_lr = ssm_p[:, 0:64] into a dense bf16 (MROWS x 64) matrix.
// ---------------------------------------------------------------------------
__global__ __launch_bounds__(256) void pack_dtlr_kernel(
    const float* __restrict__ ssm, unsigned short* __restrict__ dtlr) {
  int idx = blockIdx.x * 256 + threadIdx.x;
  if (idx >= MROWS * RR) return;
  int r = idx & (RR - 1);
  int m = idx >> 6;
  dtlr[idx] = f2bf(ssm[m * SSMW + r]);
}

// ---------------------------------------------------------------------------
// In-place dt = softplus(dt_pre + b_dt)
// ---------------------------------------------------------------------------
__global__ __launch_bounds__(256) void softplus_kernel(float* __restrict__ dt,
                                                       const float* __restrict__ bdt) {
  int idx = blockIdx.x * 256 + threadIdx.x;
  if (idx >= MROWS * DD) return;
  int d = idx & (DD - 1);
  float v = dt[idx] + bdt[d];
  dt[idx] = (v > 20.f) ? v : log1pf(__expf(v));
}

// ---------------------------------------------------------------------------
// Fused selective scan. One lane per (d,n) state; 16 lanes (one n-group)
// tree-reduce y = sum_n h*C via shfl_xor; lane n==0 applies skip + gate and
// stores bf16 y for the output GEMM. Grid: (D/16, B), block 256 (=16 d x 16 n).
// All per-step operands are staged into LDS in 64-step double-buffered chunks
// via async global->LDS DMA, so the sequential chain never touches HBM.
// ---------------------------------------------------------------------------
__global__ __launch_bounds__(256) void ssm_scan_kernel(
    const float* __restrict__ xs, const float* __restrict__ dt,
    const float* __restrict__ ssm, const float* __restrict__ proj,
    const float* __restrict__ A_log, const float* __restrict__ Dp,
    unsigned short* __restrict__ ybf) {
  __shared__ __align__(16) float s_bc[2][CHUNK * 32];  // B(16)+C(16) per step
  __shared__ __align__(16) float s_dt[2][CHUNK * 16];
  __shared__ __align__(16) float s_xs[2][CHUNK * 16];
  __shared__ __align__(16) float s_gt[2][CHUNK * 16];

  const int t   = threadIdx.x;
  const int n   = t & 15;
  const int dl  = t >> 4;
  const int b   = blockIdx.y;
  const int d0  = blockIdx.x * 16;
  const int d   = d0 + dl;
  const float Ac  = -__expf(A_log[(size_t)d * NN + n]);
  const float dpv = Dp[d];
  const size_t mb = (size_t)b * LL;

  // Issue one chunk's worth of async global->LDS copies (5 x b128 per thread).
  auto issue_chunk = [&](int c, int buf) {
    {  // B/C rows: 64 steps x 32 floats; thread copies 8 contiguous floats
      int e = t * 8;
      int st = e >> 5, j = e & 31;
      const float* g = ssm + (mb + c * CHUNK + st) * SSMW + RR + j;
      float* l = &s_bc[buf][e];
      cp16_g2l(g, l);
      cp16_g2l(g + 4, l + 4);
    }
    {  // dt / x / gate: 64 steps x 16 floats; thread copies 4 contiguous
      int e = t * 4;
      int st = e >> 4, dj = e & 15;
      size_t row = mb + c * CHUNK + st;
      cp16_g2l(dt + row * DD + d0 + dj,                &s_dt[buf][e]);
      cp16_g2l(xs + row * DD + d0 + dj,                &s_xs[buf][e]);
      cp16_g2l(proj + row * (2 * DD) + DD + d0 + dj,   &s_gt[buf][e]);
    }
  };

  float h = 0.f;
  issue_chunk(0, 0);
  for (int c = 0; c < NCHUNK; ++c) {
    // Prefetch next chunk into the other buffer (wraps harmlessly at the end:
    // it refills an already-consumed buffer that is never read again).
    issue_chunk((c + 1) & (NCHUNK - 1), (c + 1) & 1);
    wait_chunk_ready();              // s_wait_asynccnt 5 + barrier
    const int buf = c & 1;
    for (int s = 0; s < CHUNK; ++s) {
      float dtv = s_dt[buf][s * 16 + dl];
      float xv  = s_xs[buf][s * 16 + dl];
      float Bv  = s_bc[buf][s * 32 + n];
      float Cv  = s_bc[buf][s * 32 + 16 + n];
      float dA  = __expf(dtv * Ac);
      h = dA * h + dtv * Bv * xv;
      float p = h * Cv;
      p += __shfl_xor(p, 1, 32);
      p += __shfl_xor(p, 2, 32);
      p += __shfl_xor(p, 4, 32);
      p += __shfl_xor(p, 8, 32);
      if (n == 0) {
        float g  = s_gt[buf][s * 16 + dl];
        float yo = (p + xv * dpv) * (g / (1.f + __expf(-g)));
        ybf[(mb + c * CHUNK + s) * DD + d] = f2bf(yo);
      }
    }
    __syncthreads();  // everyone done reading buf before it is refilled
  }
}

// ---------------------------------------------------------------------------
extern "C" void kernel_launch(void* const* d_in, const int* in_sizes, int n_in,
                              void* d_out, int out_size, void* d_ws, size_t ws_size,
                              hipStream_t stream) {
  (void)in_sizes; (void)n_in; (void)out_size; (void)ws_size;
  const float* hidden = (const float*)d_in[0];
  const float* W_in   = (const float*)d_in[1];
  const float* conv_w = (const float*)d_in[2];
  const float* conv_b = (const float*)d_in[3];
  const float* W_x    = (const float*)d_in[4];
  const float* W_dt   = (const float*)d_in[5];
  const float* b_dt   = (const float*)d_in[6];
  const float* A_log  = (const float*)d_in[7];
  const float* D_par  = (const float*)d_in[8];
  const float* W_out  = (const float*)d_in[9];
  float* out = (float*)d_out;

  char* p = (char*)d_ws;
  auto alloc = [&](size_t bytes) -> char* {
    char* r = p;
    p += (bytes + 255) & ~size_t(255);
    return r;
  };
  unsigned short* hid_bf  = (unsigned short*)alloc((size_t)MROWS * HH * 2);
  unsigned short* Win_bf  = (unsigned short*)alloc((size_t)2 * DD * HH * 2);
  unsigned short* Wx_bf   = (unsigned short*)alloc((size_t)SSMW * DD * 2);
  unsigned short* Wdt_bf  = (unsigned short*)alloc((size_t)DD * RR * 2);
  unsigned short* Wout_bf = (unsigned short*)alloc((size_t)HH * DD * 2);
  float*          projf   = (float*)alloc((size_t)MROWS * 2 * DD * 4);
  float*          xs      = (float*)alloc((size_t)MROWS * DD * 4);
  unsigned short* xbf     = (unsigned short*)alloc((size_t)MROWS * DD * 2);
  float*          ssm     = (float*)alloc((size_t)MROWS * SSMW * 4);
  unsigned short* dtlr    = (unsigned short*)alloc((size_t)MROWS * RR * 2);
  float*          dtbuf   = (float*)alloc((size_t)MROWS * DD * 4);
  unsigned short* ybf     = (unsigned short*)alloc((size_t)MROWS * DD * 2);

  auto cvt = [&](const float* src, unsigned short* dst, int n) {
    cvt_bf16_kernel<<<(n + 255) / 256, 256, 0, stream>>>(src, dst, n);
  };

  // 1) bf16 conversions of activations + weights
  cvt(hidden, hid_bf, MROWS * HH);
  cvt(W_in,   Win_bf, 2 * DD * HH);
  cvt(W_x,    Wx_bf,  SSMW * DD);
  cvt(W_dt,   Wdt_bf, DD * RR);
  cvt(W_out,  Wout_bf, HH * DD);

  // 2) proj = hidden @ W_in^T            (2048x1024)x(4096x1024)^T  NT=4
  {
    int ngroups = (2 * DD) / 64, blocks = ngroups * (MROWS / 64);
    wmma_gemm_bf16_kernel<4><<<blocks, 128, 0, stream>>>(hid_bf, Win_bf, projf,
                                                         MROWS, 2 * DD, HH);
  }

  // 3) x = silu(depthwise_conv(proj[:, :D]) + b)
  conv_silu_kernel<<<(MROWS * DD + 255) / 256, 256, 0, stream>>>(projf, conv_w,
                                                                 conv_b, xs, xbf);

  // 4) ssm_p = x @ W_x^T                 (2048x2048)x(96x2048)^T    NT=6
  {
    int ngroups = SSMW / 96, blocks = ngroups * (MROWS / 64);
    wmma_gemm_bf16_kernel<6><<<blocks, 128, 0, stream>>>(xbf, Wx_bf, ssm,
                                                         MROWS, SSMW, DD);
  }

  // 5) dt_pre = dt_lr @ W_dt^T           (2048x64)x(2048x64)^T      NT=4
  pack_dtlr_kernel<<<(MROWS * RR + 255) / 256, 256, 0, stream>>>(ssm, dtlr);
  {
    int ngroups = DD / 64, blocks = ngroups * (MROWS / 64);
    wmma_gemm_bf16_kernel<4><<<blocks, 128, 0, stream>>>(dtlr, Wdt_bf, dtbuf,
                                                         MROWS, DD, RR);
  }

  // 6) dt = softplus(dt_pre + b_dt), in-place
  softplus_kernel<<<(MROWS * DD + 255) / 256, 256, 0, stream>>>(dtbuf, b_dt);

  // 7) fused selective scan + skip + gate -> y (bf16), async-LDS staged
  ssm_scan_kernel<<<dim3(DD / 16, BB), 256, 0, stream>>>(xs, dtbuf, ssm, projf,
                                                         A_log, D_par, ybf);

  // 8) out = y @ W_out^T                 (2048x2048)x(1024x2048)^T  NT=4
  {
    int ngroups = HH / 64, blocks = ngroups * (MROWS / 64);
    wmma_gemm_bf16_kernel<4><<<blocks, 128, 0, stream>>>(ybf, Wout_bf, out,
                                                         MROWS, HH, DD);
  }
}
